// SatGNN_36593121362096
// MI455X (gfx1250) — matrix operand order repaired and checked by reading, not compiled, not run on hardware
//
#include <hip/hip_runtime.h>
#include <math.h>

// ---------------- problem constants (match reference) ----------------
#define NV 131072
#define NC 65536
#define DD 128
#define E_NEG 131072
#define E_CON 524288
#define NL 3
#define NB 32

// ---------------- bf16 helpers / WMMA types ----------------
typedef __bf16 bf16_t;
typedef bf16_t v16bf __attribute__((ext_vector_type(16)));
typedef float  v8f   __attribute__((ext_vector_type(8)));

__device__ __forceinline__ unsigned short f2bf(float f) {
    unsigned int u = __float_as_uint(f);
    unsigned int r = u + 0x7FFFu + ((u >> 16) & 1u);   // round-to-nearest-even
    return (unsigned short)(r >> 16);
}

__device__ __forceinline__ void atomicMaxFloat(float* addr, float val) {
    // valid because all stored values come through the same scheme, init = -inf
    if (val >= 0.0f) atomicMax((int*)addr, __float_as_int(val));
    else             atomicMin((unsigned int*)addr, __float_as_uint(val));
}

__device__ __forceinline__ float leaky02(float v) {
    return v > 0.0f ? v : 0.2f * v;
}

// ---------------- operand conversion ----------------
// features: float -> bf16, 4 elements/thread (all sizes are multiples of 4)
__global__ void cvt_bf16x4(const float* __restrict__ s, unsigned short* __restrict__ d,
                           size_t n4) {
    size_t i = (size_t)blockIdx.x * blockDim.x + threadIdx.x;
    if (i >= n4) return;
    float4 f = ((const float4*)s)[i];
    ushort4 o;
    o.x = f2bf(f.x); o.y = f2bf(f.y); o.z = f2bf(f.z); o.w = f2bf(f.w);
    ((ushort4*)d)[i] = o;
}

// weights: convert + transpose nine 128x128 matrices: Wt[m][n][k] = bf16(W[m][k][n])
// grid (128, 9), block 128
__global__ void cvt_wt(const float* __restrict__ Wall, unsigned short* __restrict__ WtAll) {
    const int m = blockIdx.y;
    const int n = blockIdx.x;
    const int k = threadIdx.x;
    WtAll[((size_t)m * DD + n) * DD + k] = f2bf(Wall[((size_t)m * DD + k) * DD + n]);
}

// ---------------- GEMM: H[N x 128] = X[N x 128] @ W[128 x 128] ----------------
// Xbf: bf16 row-major; Wt: bf16, Wt[n*128 + k] = W[k][n] (K-contiguous per column)
// block = 128 threads (4 waves); each wave owns a 16-row stripe.
// A fragments for all K stay resident; col tiles iterate with one 8-VGPR accumulator.
__global__ void gemm_bf16_wmma(const unsigned short* __restrict__ Xbf,
                               const unsigned short* __restrict__ Wt,
                               float* __restrict__ H, int N) {
    const int lane = threadIdx.x & 31;
    const int wave = threadIdx.x >> 5;
    const int hlf  = lane >> 4;     // half-wave 0/1
    const int l16  = lane & 15;
    const int row0 = (blockIdx.x * 4 + wave) * 16;

    // ---- A tiles (16x32 bf16 each, ISA 7.12.2): elems 0..7 = K k0+8*hlf+i, 8..15 = +16
    union { v16bf v; uint4 q[2]; } A[4];
    const unsigned short* xr = Xbf + (size_t)(row0 + l16) * DD + 8 * hlf;
    #pragma unroll
    for (int k = 0; k < 4; ++k) {
        A[k].q[0] = *(const uint4*)(xr + 32 * k);
        A[k].q[1] = *(const uint4*)(xr + 32 * k + 16);
    }

    #pragma unroll
    for (int n = 0; n < 8; ++n) {
        // B tile n (32x16 bf16 per K-step): elems 0..7 = K k0+8*hlf+i, 8..15 = +16
        //                                          = Wt[(n*16+l16)*128 + k]
        const unsigned short* wp = Wt + (size_t)(n * 16 + l16) * DD + 8 * hlf;
        __builtin_prefetch(wp + DD, 0, 0);   // gfx1250 global_prefetch_b8
        v8f acc = {};
        #pragma unroll
        for (int k = 0; k < 4; ++k) {
            union { v16bf v; uint4 q[2]; } B;
            B.q[0] = *(const uint4*)(wp + 32 * k);
            B.q[1] = *(const uint4*)(wp + 32 * k + 16);
            acc = __builtin_amdgcn_wmma_f32_16x16x32_bf16(
                      false, A[k].v, false, B.v, (short)0, acc, false, false);
        }
        // C/D layout: VGPR j -> M = j + 8*hlf, N = l16 (within tile n)
        #pragma unroll
        for (int j = 0; j < 8; ++j)
            H[(size_t)(row0 + j + 8 * hlf) * DD + n * 16 + l16] = acc[j];
    }
}

// ---------------- out[i] = dot(H[i,:], a)  (one wave per row) ----------------
__global__ void rowdot(const float* __restrict__ H, const float* __restrict__ a,
                       float* __restrict__ out, int N) {
    const int lane = threadIdx.x & 31;
    const int row  = blockIdx.x * 4 + (threadIdx.x >> 5);
    if (row >= N) return;
    const float* h = H + (size_t)row * DD;
    float s = 0.0f;
    #pragma unroll
    for (int i = 0; i < 4; ++i) s += h[lane + 32 * i] * a[lane + 32 * i];
    #pragma unroll
    for (int off = 16; off > 0; off >>= 1) s += __shfl_xor(s, off, 32);
    if (lane == 0) out[row] = s;
}

// ---------------- wvec = Wd @ a_d  (128 outputs, 1 block of 128) ----------------
__global__ void matvec128(const float* __restrict__ Wd, const float* __restrict__ a,
                          float* __restrict__ wvec) {
    const int i = threadIdx.x;
    float s = 0.0f;
    for (int j = 0; j < DD; ++j) s += Wd[(size_t)i * DD + j] * a[j];
    wvec[i] = s;
}

// ---------------- fills / init ----------------
__global__ void fill_f32(float* p, float v, size_t n) {
    size_t i = (size_t)blockIdx.x * blockDim.x + threadIdx.x;
    if (i < n) p[i] = v;
}
__global__ void init_acc_bias(float* acc, const float* __restrict__ b0,
                              const float* __restrict__ b1, size_t n) {
    size_t i = (size_t)blockIdx.x * blockDim.x + threadIdx.x;
    if (i >= n) return;
    int d = (int)(i & (DD - 1));
    acc[i] = b0[d] + (b1 ? b1[d] : 0.0f);
}
__global__ void relu_copy(float* __restrict__ dst, const float* __restrict__ src, size_t n) {
    size_t i = (size_t)blockIdx.x * blockDim.x + threadIdx.x;
    if (i < n) dst[i] = fmaxf(src[i], 0.0f);
}

// ---------------- edge passes ----------------
__global__ void edge_max(const int* __restrict__ src, const int* __restrict__ dst,
                         const float* __restrict__ es, const float* __restrict__ ed,
                         float* m, int E) {
    int e = blockIdx.x * blockDim.x + threadIdx.x;
    if (e >= E) return;
    int d = dst[e];
    float v = leaky02(es[src[e]] + ed[d]);
    atomicMaxFloat(&m[d], v);
}
__global__ void edge_sum(const int* __restrict__ src, const int* __restrict__ dst,
                         const float* __restrict__ es, const float* __restrict__ ed,
                         const float* __restrict__ m, float* den, int E) {
    int e = blockIdx.x * blockDim.x + threadIdx.x;
    if (e >= E) return;
    int d = dst[e];
    float v = leaky02(es[src[e]] + ed[d]);
    atomicAdd(&den[d], expf(v - m[d]));
}
// one wave per edge: acc[dst,:] += hs[src,:] * alpha
__global__ void edge_aggr(const int* __restrict__ src, const int* __restrict__ dst,
                          const float* __restrict__ es, const float* __restrict__ ed,
                          const float* __restrict__ m, const float* __restrict__ den,
                          const float* __restrict__ hs, float* acc, int E) {
    const int lane = threadIdx.x & 31;
    const int e = (int)(((size_t)blockIdx.x * blockDim.x + threadIdx.x) >> 5);
    if (e >= E) return;
    const int s = src[e], d = dst[e];
    float v = leaky02(es[s] + ed[d]);
    const float alpha = expf(v - m[d]) / (den[d] + 1e-16f);
    const float* hrow = hs + (size_t)s * DD;
    float* arow = acc + (size_t)d * DD;
    #pragma unroll
    for (int i = 0; i < 4; ++i) {
        int c = lane + 32 * i;
        atomicAdd(&arow[c], hrow[c] * alpha);
    }
}

// ---------------- pooling + MLP ----------------
__global__ void pool_accum(const float* __restrict__ xv, const int* __restrict__ batch,
                           float* pooled, float* cnt, int n) {
    const int lane = threadIdx.x & 31;
    const int i = (int)(((size_t)blockIdx.x * blockDim.x + threadIdx.x) >> 5);
    if (i >= n) return;
    const int b = batch[i];
    if (lane == 0) atomicAdd(&cnt[b], 1.0f);
    const float* row = xv + (size_t)i * DD;
    float* p = pooled + (size_t)b * DD;
    #pragma unroll
    for (int k = 0; k < 4; ++k) atomicAdd(&p[lane + 32 * k], row[lane + 32 * k]);
}
__global__ void mlp1(const float* __restrict__ pooled, const float* __restrict__ cnt,
                     const float* __restrict__ w1, const float* __restrict__ b1,
                     float* h) {
    const int b = blockIdx.x, j = threadIdx.x;
    const float inv = 1.0f / fmaxf(cnt[b], 1.0f);
    float s = b1[j];
    for (int k = 0; k < DD; ++k)
        s += pooled[(size_t)b * DD + k] * inv * w1[(size_t)k * DD + j];
    h[(size_t)b * DD + j] = fmaxf(s, 0.0f);
}
__global__ void mlp2(const float* __restrict__ h, const float* __restrict__ w2,
                     const float* __restrict__ b2, float* out) {
    __shared__ float sb[DD];
    const int b = blockIdx.x, j = threadIdx.x;
    sb[j] = h[(size_t)b * DD + j] * w2[j];
    __syncthreads();
    for (int s = 64; s > 0; s >>= 1) {
        if (j < s) sb[j] += sb[j + s];
        __syncthreads();
    }
    if (j == 0) out[b] = sb[0] + b2[0];
}

// ---------------- host-side helpers ----------------
static inline dim3 cdiv1(size_t n, int t) { return dim3((unsigned)((n + t - 1) / t)); }

struct Scratch {
    float *xv, *xc, *acc_v, *acc_c, *hs, *es, *ed, *m, *den, *wvec, *pooled, *cnt, *hmlp;
    unsigned short *xv_bf, *xc_bf, *wt_all;   // bf16 operands
};

static void run_gat(const unsigned short* xs_bf, int Ns, const float* xd, int Nd,
                    const int* src, const int* dst, int E,
                    const unsigned short* Wt, const float* Wd,
                    const float* a_s, const float* a_d,
                    const Scratch& S, float* acc, hipStream_t st) {
    // hs = xs @ Ws  (bf16 WMMA, f32 accumulate)
    gemm_bf16_wmma<<<cdiv1(Ns, 64), 128, 0, st>>>(xs_bf, Wt, S.hs, Ns);
    // es = hs @ a_s
    rowdot<<<cdiv1(Ns, 4), 128, 0, st>>>(S.hs, a_s, S.es, Ns);
    // wvec = Wd @ a_d ;  ed = xd @ wvec   (hd only feeds the attention scalar)
    matvec128<<<1, 128, 0, st>>>(Wd, a_d, S.wvec);
    rowdot<<<cdiv1(Nd, 4), 128, 0, st>>>(xd, S.wvec, S.ed, Nd);
    // per-destination softmax
    fill_f32<<<cdiv1(Nd, 256), 256, 0, st>>>(S.m, -INFINITY, (size_t)Nd);
    fill_f32<<<cdiv1(Nd, 256), 256, 0, st>>>(S.den, 0.0f, (size_t)Nd);
    edge_max<<<cdiv1(E, 256), 256, 0, st>>>(src, dst, S.es, S.ed, S.m, E);
    edge_sum<<<cdiv1(E, 256), 256, 0, st>>>(src, dst, S.es, S.ed, S.m, S.den, E);
    edge_aggr<<<cdiv1((size_t)E * 32, 256), 256, 0, st>>>(src, dst, S.es, S.ed, S.m, S.den,
                                                          S.hs, acc, E);
}

extern "C" void kernel_launch(void* const* d_in, const int* in_sizes, int n_in,
                              void* d_out, int out_size, void* d_ws, size_t ws_size,
                              hipStream_t stream) {
    (void)in_sizes; (void)n_in; (void)out_size; (void)ws_size;
    const float* x_var   = (const float*)d_in[0];
    const float* x_con   = (const float*)d_in[1];
    const float* W_src   = (const float*)d_in[2];   // [L,3,D,D]
    const float* W_dst   = (const float*)d_in[3];
    const float* att_src = (const float*)d_in[4];   // [L,3,D]
    const float* att_dst = (const float*)d_in[5];
    const float* bias    = (const float*)d_in[6];   // [L,3,D]
    const float* mlp_w1  = (const float*)d_in[7];
    const float* mlp_b1  = (const float*)d_in[8];
    const float* mlp_w2  = (const float*)d_in[9];
    const float* mlp_b2  = (const float*)d_in[10];
    const int* en_src = (const int*)d_in[11];
    const int* en_dst = (const int*)d_in[12];
    const int* ec_src = (const int*)d_in[13];
    const int* ec_dst = (const int*)d_in[14];
    const int* batch  = (const int*)d_in[15];
    float* out = (float*)d_out;

    // ---- carve workspace ----
    float* p = (float*)d_ws;
    Scratch S;
    S.xv = p;     p += (size_t)NV * DD;
    S.xc = p;     p += (size_t)NC * DD;
    S.acc_v = p;  p += (size_t)NV * DD;
    S.acc_c = p;  p += (size_t)NC * DD;
    S.hs = p;     p += (size_t)NV * DD;     // shared by all relations (max N = NV)
    S.es = p;     p += NV;
    S.ed = p;     p += NV;
    S.m = p;      p += NV;
    S.den = p;    p += NV;
    S.wvec = p;   p += DD;
    S.pooled = p; p += (size_t)NB * DD;
    S.cnt = p;    p += NB;
    S.hmlp = p;   p += (size_t)NB * DD;
    unsigned short* q = (unsigned short*)p;
    S.xv_bf = q;  q += (size_t)NV * DD;
    S.xc_bf = q;  q += (size_t)NC * DD;
    S.wt_all = q; q += (size_t)NL * 3 * DD * DD;

    // node features (do not mutate inputs)
    hipMemcpyAsync(S.xv, x_var, (size_t)NV * DD * sizeof(float), hipMemcpyDeviceToDevice, stream);
    hipMemcpyAsync(S.xc, x_con, (size_t)NC * DD * sizeof(float), hipMemcpyDeviceToDevice, stream);

    // all nine W_src matrices -> bf16, transposed (K-contiguous per output column)
    cvt_wt<<<dim3(DD, NL * 3), DD, 0, stream>>>(W_src, S.wt_all);

    const size_t WSTRIDE = (size_t)DD * DD;
    const size_t ASTRIDE = DD;

    for (int l = 0; l < NL; ++l) {
        const size_t r0 = (size_t)(l * 3 + 0), r1 = (size_t)(l * 3 + 1), r2 = (size_t)(l * 3 + 2);

        // bf16 copies of current features (GEMM A operands)
        cvt_bf16x4<<<cdiv1((size_t)NV * DD / 4, 256), 256, 0, stream>>>(
            S.xv, S.xv_bf, (size_t)NV * DD / 4);
        cvt_bf16x4<<<cdiv1((size_t)NC * DD / 4, 256), 256, 0, stream>>>(
            S.xc, S.xc_bf, (size_t)NC * DD / 4);

        // accumulators seeded with biases (HeteroConv sum)
        init_acc_bias<<<cdiv1((size_t)NV * DD, 256), 256, 0, stream>>>(
            S.acc_v, bias + r0 * ASTRIDE, bias + r2 * ASTRIDE, (size_t)NV * DD);
        init_acc_bias<<<cdiv1((size_t)NC * DD, 256), 256, 0, stream>>>(
            S.acc_c, bias + r1 * ASTRIDE, nullptr, (size_t)NC * DD);

        // rel 0: var -> var over negation edges
        run_gat(S.xv_bf, NV, S.xv, NV, en_src, en_dst, E_NEG,
                S.wt_all + r0 * WSTRIDE, W_dst + r0 * WSTRIDE,
                att_src + r0 * ASTRIDE, att_dst + r0 * ASTRIDE, S, S.acc_v, stream);
        // rel 2: con -> var (reversed connection edges)
        run_gat(S.xc_bf, NC, S.xv, NV, ec_dst, ec_src, E_CON,
                S.wt_all + r2 * WSTRIDE, W_dst + r2 * WSTRIDE,
                att_src + r2 * ASTRIDE, att_dst + r2 * ASTRIDE, S, S.acc_v, stream);
        // rel 1: var -> con
        run_gat(S.xv_bf, NV, S.xc, NC, ec_src, ec_dst, E_CON,
                S.wt_all + r1 * WSTRIDE, W_dst + r1 * WSTRIDE,
                att_src + r1 * ASTRIDE, att_dst + r1 * ASTRIDE, S, S.acc_c, stream);

        // relu updates (after all relations consumed old xv/xc)
        relu_copy<<<cdiv1((size_t)NV * DD, 256), 256, 0, stream>>>(S.xv, S.acc_v, (size_t)NV * DD);
        relu_copy<<<cdiv1((size_t)NC * DD, 256), 256, 0, stream>>>(S.xc, S.acc_c, (size_t)NC * DD);
    }

    // global mean pool over variable nodes, then vote MLP
    fill_f32<<<cdiv1((size_t)NB * DD, 256), 256, 0, stream>>>(S.pooled, 0.0f, (size_t)NB * DD);
    fill_f32<<<1, NB, 0, stream>>>(S.cnt, 0.0f, (size_t)NB);
    pool_accum<<<cdiv1((size_t)NV * 32, 256), 256, 0, stream>>>(S.xv, batch, S.pooled, S.cnt, NV);
    mlp1<<<NB, DD, 0, stream>>>(S.pooled, S.cnt, mlp_w1, mlp_b1, S.hmlp);
    mlp2<<<NB, DD, 0, stream>>>(S.hmlp, mlp_w2, mlp_b2, out);
}
